// Olmo3_5HybridGatedDeltaNet_68341519614278
// MI455X (gfx1250) — matrix operand order repaired and checked by reading, compiled, not run on hardware
//
#include <hip/hip_runtime.h>
#include <hip/hip_bf16.h>

// ---------------------------------------------------------------------------
// Gated DeltaNet (chunked) for MI455X / gfx1250, wave32, WMMA bf16.
// Grid: B*H*2 workgroups (Dv split in halves), 256 threads (8 waves) each.
// All WMMA operand buffers are laid out so every fragment load is two
// 16-byte ds_load_b128 per lane (A: row-major, B: stored transposed).
// ---------------------------------------------------------------------------

typedef __bf16 bf;
typedef __attribute__((ext_vector_type(16))) __bf16 v16bf;
typedef __attribute__((ext_vector_type(8)))  __bf16 v8bf;
typedef __attribute__((ext_vector_type(4)))  __bf16 v4bf;
typedef __attribute__((ext_vector_type(8)))  float  v8f;

#define BD   2
#define SD   4096
#define HD   16
#define DKD  128
#define DVD  128
#define CCH  64
#define NCH  (SD / CCH)
#define DVS  2
#define DVH  (DVD / DVS)       // 64 value-columns per workgroup

// LDS carve (bytes). Must match device-side carving below.
static constexpr int LDS_BYTES_I =
      8192 * 4              /* fs: f32 staging / kk / qk scratch              */
    + 7 * (8192 * 2)        /* q, qg, k, kb, kbgT, kdT, kcd                   */
    + 2 * (4096 * 2)        /* vbT, vnT                                       */
    + 10 * (4096 * 2)       /* P,Pt,PT,PtT,Q,Qt,QT,QtT,R,attn (64x64)         */
    + 8192 * 2              /* ST: state mirror, transposed [Dv][Dk]          */
    + 5 * 64 * 4            /* gcv, egv, erv, btv, rnv                        */
    + 256 * 4;              /* prt partial sums                               */

// ---- WMMA fragment loaders (layouts per CDNA5 ISA 7.12.2, wave32) ---------

__device__ __forceinline__ v16bf cat8(v8bf lo, v8bf hi) {
  return __builtin_shufflevector(lo, hi, 0, 1, 2, 3, 4, 5, 6, 7,
                                         8, 9, 10, 11, 12, 13, 14, 15);
}

// A fragment: 16x32 bf16 tile from row-major A (lda elems). Per lane:
// f[0..7] = row[K=kb+0..7], f[8..15] = row[K=16+kb..16+kb+7]  (kb = 0|8)
__device__ __forceinline__ v16bf ldA(const bf* A, int lda, int m0, int k0, int lane) {
  const bf* row = A + (size_t)(m0 + (lane & 15)) * lda + k0 + ((lane & 16) ? 8 : 0);
  v8bf lo = *(const v8bf*)(row);
  v8bf hi = *(const v8bf*)(row + 16);
  return cat8(lo, hi);
}

// B fragment: 32x16 tile where B = M^T, M row-major [N][K] (ldm elems):
// B[k][n] = M[n][k]. Per lane: f[0..7] = K=kb+0..7, f[8..15] = K=16+kb..16+kb+7 (kb = 0|8) at row n.
__device__ __forceinline__ v16bf ldBT(const bf* M, int ldm, int k0, int n0, int lane) {
  const bf* row = M + (size_t)(n0 + (lane & 15)) * ldm + k0 + ((lane & 16) ? 8 : 0);
  v8bf lo = *(const v8bf*)(row);
  v8bf hi = *(const v8bf*)(row + 16);
  return cat8(lo, hi);
}

__device__ __forceinline__ v8f wmma_bf(v16bf a, v16bf b, v8f c) {
  return __builtin_amdgcn_wmma_f32_16x16x32_bf16(false, a, false, b, (short)0, c,
                                                 false, false);
}

// C/D element r lives at (m0 + r + 8*(lane>=16), n0 + (lane&15)).
__device__ __forceinline__ void stC_f32(float* C, int ldc, int m0, int n0, int lane, v8f a) {
  int mrow = m0 + ((lane >> 4) << 3);
  int ncol = n0 + (lane & 15);
#pragma unroll
  for (int r = 0; r < 8; ++r) C[(mrow + r) * ldc + ncol] = a[r];
}

__device__ __forceinline__ void stC_bf(bf* C, int ldc, int m0, int n0, int lane, v8f a) {
  int mrow = m0 + ((lane >> 4) << 3);
  int ncol = n0 + (lane & 15);
#pragma unroll
  for (int r = 0; r < 8; ++r) C[(mrow + r) * ldc + ncol] = (bf)a[r];
}

// Transposed C store: 8 per-lane elements are contiguous -> one b128 store.
__device__ __forceinline__ void stC_bfT(bf* CT, int ldt, int m0, int n0, int lane, v8f a) {
  int mrow = m0 + ((lane >> 4) << 3);
  int ncol = n0 + (lane & 15);
  v8bf t;
#pragma unroll
  for (int r = 0; r < 8; ++r) t[r] = (bf)a[r];
  *(v8bf*)(CT + (size_t)ncol * ldt + mrow) = t;
}

// ---------------------------------------------------------------------------

__global__ __launch_bounds__(256, 1)
void gdn_chunk_kernel(const float* __restrict__ qG, const float* __restrict__ kG,
                      const float* __restrict__ vG, const float* __restrict__ gG,
                      const float* __restrict__ bG, float* __restrict__ oG) {
  extern __shared__ char smem_raw[];
  char* p = smem_raw;
  float* fs    = (float*)p; p += 8192 * 4;   // staging + kk/qk scratch
  bf* q_bf     = (bf*)p;    p += 8192 * 2;   // q * invnorm * 1/sqrt(Dk)   [64][128]
  bf* qg_bf    = (bf*)p;    p += 8192 * 2;   // q_bf * exp(gc)             [64][128]
  bf* k_bf     = (bf*)p;    p += 8192 * 2;   // k * invnorm                [64][128]
  bf* kb_bf    = (bf*)p;    p += 8192 * 2;   // k_bf * beta                [64][128]
  bf* kbgT     = (bf*)p;    p += 8192 * 2;   // (k*beta*exp(gc))^T         [128][64]
  bf* kdT_bf   = (bf*)p;    p += 8192 * 2;   // (k * exp(g_last - gc))^T   [128][64]
  bf* kcd_bf   = (bf*)p;    p += 8192 * 2;   // -(Tm @ kbg)                [64][128]
  bf* vbT      = (bf*)p;    p += 4096 * 2;   // (v*beta)^T                 [64][64]
  bf* vnT      = (bf*)p;    p += 4096 * 2;   // v_new^T                    [64][64]
  bf* Pb       = (bf*)p;    p += 4096 * 2;   // P row-major (A operand)
  bf* Pt       = (bf*)p;    p += 4096 * 2;
  bf* PbT      = (bf*)p;    p += 4096 * 2;   // P^T (fast elementwise read)
  bf* PtT      = (bf*)p;    p += 4096 * 2;
  bf* Qb       = (bf*)p;    p += 4096 * 2;   // Q row-major (A operand)
  bf* Qt       = (bf*)p;    p += 4096 * 2;
  bf* QbT      = (bf*)p;    p += 4096 * 2;   // Q^T (B operand)
  bf* QtT      = (bf*)p;    p += 4096 * 2;
  bf* Rb       = (bf*)p;    p += 4096 * 2;   // R = P@Q scratch (A operand)
  bf* at_bf    = (bf*)p;    p += 4096 * 2;   // masked decayed attn (A op)
  bf* ST       = (bf*)p;    p += 8192 * 2;   // state^T                    [64][128]
  float* gcv   = (float*)p; p += 64 * 4;     // cumsum(g) within chunk
  float* egv   = (float*)p; p += 64 * 4;     // exp(gc)
  float* erv   = (float*)p; p += 64 * 4;     // exp(g_last - gc)
  float* btv   = (float*)p; p += 64 * 4;     // beta
  float* rnv   = (float*)p; p += 64 * 4;     // 1/sqrt(sumsq + eps)
  float* prt   = (float*)p; p += 256 * 4;    // sumsq partials

  const int tid  = threadIdx.x;
  const int lane = tid & 31;
  const int wave = tid >> 5;

  const int wg   = blockIdx.x;               // [0, B*H*DVS)
  const int b    = wg / (HD * DVS);
  const int rem  = wg % (HD * DVS);
  const int h    = rem / DVS;
  const int dv0  = (rem % DVS) * DVH;

  const float qscale = 0.08838834764831845f; // 1/sqrt(128)

  // state: wave w owns Dk rows [16w, 16w+16), 4 N-tiles of 16 -> 4 C frags
  v8f Sacc[4];
#pragma unroll
  for (int t = 0; t < 4; ++t) Sacc[t] = (v8f){0, 0, 0, 0, 0, 0, 0, 0};
  for (int i = tid; i < DKD * DVH; i += 256) ST[i] = (bf)0.0f;
  __syncthreads();

  for (int ch = 0; ch < NCH; ++ch) {
    const int s0 = ch * CCH;

    // ---- gates / beta ------------------------------------------------------
    if (tid < CCH) {
      float gv = gG[((size_t)b * SD + s0 + tid) * HD + h];
      gcv[tid] = fminf(fmaxf(gv, -20.f), 20.f);
      btv[tid] = bG[((size_t)b * SD + s0 + tid) * HD + h];
    }
    __syncthreads();
    if (tid == 0) {
      float a = 0.f;
      for (int i = 0; i < CCH; ++i) { a += gcv[i]; gcv[i] = a; }
    }
    __syncthreads();
    if (tid < CCH) {
      egv[tid] = __expf(gcv[tid]);
      erv[tid] = __expf(gcv[CCH - 1] - gcv[tid]);
    }

    // ---- stage + normalize q  (float4 staging) ----------------------------
    for (int i4 = tid; i4 < CCH * DKD / 4; i4 += 256) {
      int r = i4 >> 5, c4 = i4 & 31;
      size_t base = (((size_t)b * SD + s0 + r) * HD + h) * (size_t)DKD;
      float4 x = *(const float4*)(qG + base + c4 * 4);
      ((float4*)fs)[i4] = x;
      if ((ch + 1 < NCH) && ((i4 & 3) == 0))
        __builtin_prefetch(qG + base + (size_t)CCH * HD * DKD + c4 * 4, 0, 0);
    }
    __syncthreads();
    {
      int r = tid >> 2, part = tid & 3;
      const float* row = fs + r * DKD + part * 32;
      float s = 0.f;
#pragma unroll
      for (int i = 0; i < 32; ++i) s += row[i] * row[i];
      prt[tid] = s;
    }
    __syncthreads();
    if (tid < CCH)
      rnv[tid] = rsqrtf(prt[4 * tid] + prt[4 * tid + 1] + prt[4 * tid + 2] +
                        prt[4 * tid + 3] + 1e-6f);
    __syncthreads();
    for (int i4 = tid; i4 < CCH * DKD / 4; i4 += 256) {
      int r = i4 >> 5;
      float4 x = ((const float4*)fs)[i4];
      float rn = rnv[r] * qscale, eg = egv[r];
      v4bf qv = { (bf)(x.x * rn), (bf)(x.y * rn), (bf)(x.z * rn), (bf)(x.w * rn) };
      v4bf gv = { (bf)(x.x * rn * eg), (bf)(x.y * rn * eg),
                  (bf)(x.z * rn * eg), (bf)(x.w * rn * eg) };
      *(v4bf*)(q_bf  + i4 * 4) = qv;
      *(v4bf*)(qg_bf + i4 * 4) = gv;
    }
    __syncthreads();

    // ---- stage + normalize k, derive kb / kbg^T / kdec^T ------------------
    for (int i4 = tid; i4 < CCH * DKD / 4; i4 += 256) {
      int r = i4 >> 5, c4 = i4 & 31;
      size_t base = (((size_t)b * SD + s0 + r) * HD + h) * (size_t)DKD;
      float4 x = *(const float4*)(kG + base + c4 * 4);
      ((float4*)fs)[i4] = x;
      if ((ch + 1 < NCH) && ((i4 & 3) == 0))
        __builtin_prefetch(kG + base + (size_t)CCH * HD * DKD + c4 * 4, 0, 0);
    }
    __syncthreads();
    {
      int r = tid >> 2, part = tid & 3;
      const float* row = fs + r * DKD + part * 32;
      float s = 0.f;
#pragma unroll
      for (int i = 0; i < 32; ++i) s += row[i] * row[i];
      prt[tid] = s;
    }
    __syncthreads();
    if (tid < CCH)
      rnv[tid] = rsqrtf(prt[4 * tid] + prt[4 * tid + 1] + prt[4 * tid + 2] +
                        prt[4 * tid + 3] + 1e-6f);
    __syncthreads();
    for (int i4 = tid; i4 < CCH * DKD / 4; i4 += 256) {
      int r = i4 >> 5, d = (i4 & 31) * 4;
      float4 x = ((const float4*)fs)[i4];
      float rn = rnv[r], bt = btv[r], eg = egv[r], er = erv[r];
      float kn0 = x.x * rn, kn1 = x.y * rn, kn2 = x.z * rn, kn3 = x.w * rn;
      v4bf kv = { (bf)kn0, (bf)kn1, (bf)kn2, (bf)kn3 };
      v4bf bv = { (bf)(kn0 * bt), (bf)(kn1 * bt), (bf)(kn2 * bt), (bf)(kn3 * bt) };
      *(v4bf*)(k_bf  + i4 * 4) = kv;
      *(v4bf*)(kb_bf + i4 * 4) = bv;
      float bg = bt * eg;
      kbgT[(d + 0) * CCH + r] = (bf)(kn0 * bg);
      kbgT[(d + 1) * CCH + r] = (bf)(kn1 * bg);
      kbgT[(d + 2) * CCH + r] = (bf)(kn2 * bg);
      kbgT[(d + 3) * CCH + r] = (bf)(kn3 * bg);
      kdT_bf[(d + 0) * CCH + r] = (bf)(kn0 * er);
      kdT_bf[(d + 1) * CCH + r] = (bf)(kn1 * er);
      kdT_bf[(d + 2) * CCH + r] = (bf)(kn2 * er);
      kdT_bf[(d + 3) * CCH + r] = (bf)(kn3 * er);
    }
    __syncthreads();

    // ---- (v * beta)^T for this Dv half ------------------------------------
    for (int i4 = tid; i4 < CCH * DVH / 4; i4 += 256) {
      int r = i4 >> 4, dl = (i4 & 15) * 4;
      float4 x = *(const float4*)(vG + (((size_t)b * SD + s0 + r) * HD + h) * (size_t)DVD
                                   + dv0 + dl);
      float bt = btv[r];
      vbT[(dl + 0) * CCH + r] = (bf)(x.x * bt);
      vbT[(dl + 1) * CCH + r] = (bf)(x.y * bt);
      vbT[(dl + 2) * CCH + r] = (bf)(x.z * bt);
      vbT[(dl + 3) * CCH + r] = (bf)(x.w * bt);
    }

    // ---- kk = kb @ k^T and qk = q @ k^T  (both 64x64, K=128) --------------
    float* kkf = fs;          // [64][64]
    float* qkf = fs + 4096;   // [64][64]
    for (int job = wave; job < 32; job += 8) {
      int t = job & 15, mt = t >> 2, nt = t & 3;
      const bf* Ap = (job < 16) ? kb_bf : q_bf;
      v8f acc = {};
#pragma unroll
      for (int k0 = 0; k0 < DKD; k0 += 32)
        acc = wmma_bf(ldA(Ap, DKD, mt * 16, k0, lane),
                      ldBT(k_bf, DKD, k0, nt * 16, lane), acc);
      stC_f32((job < 16) ? kkf : qkf, 64, mt * 16, nt * 16, lane, acc);
    }
    __syncthreads();

    // ---- build A (strict lower): Q=A (dual), P=I+A (dual), masked attn ----
    for (int i = tid; i < CCH * CCH; i += 256) {
      int ii = i >> 6, jj = i & 63;
      float dec = __expf(gcv[ii] - gcv[jj]);
      float a   = (jj < ii) ? (-kkf[i] * dec) : 0.f;
      float pv  = a + ((ii == jj) ? 1.f : 0.f);
      Qb[i]  = (bf)a;   QbT[jj * 64 + ii] = (bf)a;
      Pb[i]  = (bf)pv;  PbT[jj * 64 + ii] = (bf)pv;
      at_bf[i] = (bf)((jj <= ii) ? (qkf[i] * dec) : 0.f);
    }
    __syncthreads();

    // ---- Tm = (I-A)^{-1} = prod_k (I + A^(2^k)) via 5 doubling steps ------
    bf* Pc = Pb;  bf* PcT = PbT; bf* Pn = Pt;  bf* PnT = PtT;
    bf* Qc = Qb;  bf* QcT = QbT; bf* Qn = Qt;  bf* QnT = QtT;
    for (int it = 0; it < 5; ++it) {
      // phase 1: Qn = Qc@Qc (dual store) ; R = Pc@Qc
      for (int job = wave; job < 32; job += 8) {
        int t = job & 15, mt = t >> 2, nt = t & 3;
        const bf* Ap = (job < 16) ? Qc : Pc;
        v8f acc = {};
#pragma unroll
        for (int k0 = 0; k0 < 64; k0 += 32)
          acc = wmma_bf(ldA(Ap, 64, mt * 16, k0, lane),
                        ldBT(QcT, 64, k0, nt * 16, lane), acc);
        if (job < 16) {
          stC_bf(Qn, 64, mt * 16, nt * 16, lane, acc);
          stC_bfT(QnT, 64, mt * 16, nt * 16, lane, acc);
        } else {
          stC_bf(Rb, 64, mt * 16, nt * 16, lane, acc);
        }
      }
      __syncthreads();
      // phase 2: Pn = Pc + R@Qc   ( = Pc (I + Qc^2) ), dual store
      for (int job = wave; job < 16; job += 8) {
        int mt = job >> 2, nt = job & 3;
        v8f acc = {};
#pragma unroll
        for (int k0 = 0; k0 < 64; k0 += 32)
          acc = wmma_bf(ldA(Rb, 64, mt * 16, k0, lane),
                        ldBT(QcT, 64, k0, nt * 16, lane), acc);
        int mrow = mt * 16 + ((lane >> 4) << 3);
        int ncol = nt * 16 + (lane & 15);
        v8bf pold = *(const v8bf*)(PcT + ncol * 64 + mrow);   // contiguous
        v8bf tnew;
#pragma unroll
        for (int r = 0; r < 8; ++r) {
          float val = acc[r] + (float)pold[r];
          Pn[(mrow + r) * 64 + ncol] = (bf)val;
          tnew[r] = (bf)val;
        }
        *(v8bf*)(PnT + ncol * 64 + mrow) = tnew;
      }
      __syncthreads();
      bf* t1;
      t1 = Pc;  Pc  = Pn;  Pn  = t1;
      t1 = PcT; PcT = PnT; PnT = t1;
      t1 = Qc;  Qc  = Qn;  Qn  = t1;
      t1 = QcT; QcT = QnT; QnT = t1;
    }
    const bf* Tm = Pc;   // row-major, unit-lower-triangular inverse

    // ---- kcd = -(Tm @ kbg)  [64x128] (A operand downstream) ---------------
    for (int job = wave; job < 32; job += 8) {
      int mt = job >> 3, nt = job & 7;
      v8f acc = {};
#pragma unroll
      for (int k0 = 0; k0 < 64; k0 += 32)
        acc = wmma_bf(ldA(Tm, 64, mt * 16, k0, lane),
                      ldBT(kbgT, CCH, k0, nt * 16, lane), acc);
      int mrow = mt * 16 + ((lane >> 4) << 3);
      int ncol = nt * 16 + (lane & 15);
#pragma unroll
      for (int r = 0; r < 8; ++r) kcd_bf[(mrow + r) * DKD + ncol] = (bf)(-acc[r]);
    }
    __syncthreads();

    // ---- v_new = Tm@vb + kcd@S   (kcd pre-negated => minus v_prime) -------
    for (int job = wave; job < 16; job += 8) {
      int mt = job >> 2, nt = job & 3;
      v8f acc = {};
#pragma unroll
      for (int k0 = 0; k0 < 64; k0 += 32)
        acc = wmma_bf(ldA(Tm, 64, mt * 16, k0, lane),
                      ldBT(vbT, CCH, k0, nt * 16, lane), acc);
#pragma unroll
      for (int k0 = 0; k0 < DKD; k0 += 32)
        acc = wmma_bf(ldA(kcd_bf, DKD, mt * 16, k0, lane),
                      ldBT(ST, DKD, k0, nt * 16, lane), acc);
      stC_bfT(vnT, CCH, mt * 16, nt * 16, lane, acc);
    }
    __syncthreads();

    // ---- out = qg@S + attn@v_new  -> global f32 ---------------------------
    for (int job = wave; job < 16; job += 8) {
      int mt = job >> 2, nt = job & 3;
      v8f acc = {};
#pragma unroll
      for (int k0 = 0; k0 < DKD; k0 += 32)
        acc = wmma_bf(ldA(qg_bf, DKD, mt * 16, k0, lane),
                      ldBT(ST, DKD, k0, nt * 16, lane), acc);
#pragma unroll
      for (int k0 = 0; k0 < 64; k0 += 32)
        acc = wmma_bf(ldA(at_bf, 64, mt * 16, k0, lane),
                      ldBT(vnT, CCH, k0, nt * 16, lane), acc);
      int mrow = mt * 16 + ((lane >> 4) << 3);
      int ncol = nt * 16 + (lane & 15);
#pragma unroll
      for (int r = 0; r < 8; ++r)
        oG[(((size_t)b * SD + s0 + mrow + r) * HD + h) * (size_t)DVD + dv0 + ncol] = acc[r];
    }

    // ---- state update: S = S*exp(g_last) + kdec^T @ v_new -----------------
    {
      float el = __expf(gcv[CCH - 1]);
#pragma unroll
      for (int nt = 0; nt < 4; ++nt) {
#pragma unroll
        for (int r = 0; r < 8; ++r) Sacc[nt][r] *= el;
#pragma unroll
        for (int k0 = 0; k0 < 64; k0 += 32)
          Sacc[nt] = wmma_bf(ldA(kdT_bf, CCH, wave * 16, k0, lane),
                             ldBT(vnT, CCH, k0, nt * 16, lane), Sacc[nt]);
      }
    }
    __syncthreads();   // all readers of ST done before overwrite
#pragma unroll
    for (int nt = 0; nt < 4; ++nt)
      stC_bfT(ST, DKD, wave * 16, nt * 16, lane, Sacc[nt]);   // one b128/lane
    __syncthreads();
  }
}

// ---------------------------------------------------------------------------

extern "C" void kernel_launch(void* const* d_in, const int* in_sizes, int n_in,
                              void* d_out, int out_size, void* d_ws, size_t ws_size,
                              hipStream_t stream) {
  (void)in_sizes; (void)n_in; (void)out_size; (void)d_ws; (void)ws_size;
  const float* q  = (const float*)d_in[0];
  const float* k  = (const float*)d_in[1];
  const float* v  = (const float*)d_in[2];
  const float* g  = (const float*)d_in[3];
  const float* be = (const float*)d_in[4];
  float* out = (float*)d_out;

  // CDNA5 allows up to 320KB LDS per workgroup; opt in past the 64KB default.
  (void)hipFuncSetAttribute(reinterpret_cast<const void*>(&gdn_chunk_kernel),
                            hipFuncAttributeMaxDynamicSharedMemorySize,
                            LDS_BYTES_I);

  dim3 grid(BD * HD * DVS);   // 64 workgroups
  dim3 block(256);            // 8 waves (wave32)
  gdn_chunk_kernel<<<grid, block, (size_t)LDS_BYTES_I, stream>>>(q, k, v, g, be, out);
}
